// PairwiseAttentionTerminal_66984309948707
// MI455X (gfx1250) — matrix operand order repaired and checked by reading, not compiled, not run on hardware
//
#include <hip/hip_runtime.h>
#include <hip/hip_bf16.h>

// Problem dims (fixed by the reference)
#define L_DIM 1024
#define B_DIM 8
#define F_DIM 256
#define H_DIM 8
#define C_DIM 32
#define HC    256              // H*C
#define ROWS  (L_DIM * B_DIM)  // 8192 token rows, row = l*B + b
#define EPS_LN 1e-5f

typedef unsigned short ushort_t;
typedef __attribute__((ext_vector_type(16))) __bf16 v16bf;
typedef __attribute__((ext_vector_type(8)))  __bf16 v8bf;
typedef __attribute__((ext_vector_type(8)))  float  v8f;
typedef __attribute__((ext_vector_type(4)))  int    v4i;

// ------------------------------------------------- async-to-LDS detection ---
#if defined(__has_builtin)
#if __has_builtin(__builtin_amdgcn_global_load_async_to_lds_b128)
#define HAVE_ASYNC 1
#endif
#endif
#ifndef HAVE_ASYNC
#define HAVE_ASYNC 0
#endif

#if HAVE_ASYNC
typedef __attribute__((address_space(3))) v4i* lds_v4ip;
typedef __attribute__((address_space(1))) v4i* gbl_v4ip;
// generic LDS pointer: addr[31:0] is the LDS offset (ISA aperture rule), so an
// integer truncation round-trip builds the addrspace(3) pointer portably.
__device__ inline lds_v4ip to_lds(void* p) {
  return (lds_v4ip)(unsigned)(unsigned long long)p;
}
__device__ inline gbl_v4ip to_global(const void* p) {
  return (gbl_v4ip)(unsigned long long)p;
}
#endif

// ---------------------------------------------------------------- helpers ---

__device__ inline ushort_t f2bf(float f) {
  unsigned u = __float_as_uint(f);
  u += 0x7FFFu + ((u >> 16) & 1u);          // round-to-nearest-even
  return (ushort_t)(u >> 16);
}

// Build a 16-element bf16 fragment from two contiguous 8-half (16B) runs.
__device__ inline v16bf make_frag(const ushort_t* p0, const ushort_t* p1) {
  uint4 a = *(const uint4*)p0;
  uint4 b = *(const uint4*)p1;
  v8bf lo = __builtin_bit_cast(v8bf, a);
  v8bf hi = __builtin_bit_cast(v8bf, b);
  v16bf r;
#pragma unroll
  for (int i = 0; i < 8; ++i) { r[i] = lo[i]; r[i + 8] = hi[i]; }
  return r;
}

__device__ inline v8f wmma_bf16(v16bf a, v16bf b, v8f c) {
  return __builtin_amdgcn_wmma_f32_16x16x32_bf16(
      false, a, false, b, (short)0, c, false, false);
}

// reductions confined to a 16-lane half (matches C/D layout)
__device__ inline float halfmax(float v) {
  v = fmaxf(v, __shfl_xor(v, 1, 32));
  v = fmaxf(v, __shfl_xor(v, 2, 32));
  v = fmaxf(v, __shfl_xor(v, 4, 32));
  v = fmaxf(v, __shfl_xor(v, 8, 32));
  return v;
}
__device__ inline float halfsum(float v) {
  v += __shfl_xor(v, 1, 32);
  v += __shfl_xor(v, 2, 32);
  v += __shfl_xor(v, 4, 32);
  v += __shfl_xor(v, 8, 32);
  return v;
}

// ---------------------------------------------------- K0: weight transpose ---
__global__ __launch_bounds__(256)
void wtrans_kernel(const float* __restrict__ W, ushort_t* __restrict__ Wt,
                   int K, int N, int Npad) {
  int idx = blockIdx.x * blockDim.x + threadIdx.x;
  if (idx >= Npad * K) return;
  int n = idx / K, k = idx % K;
  float v = (n < N) ? W[k * N + n] : 0.f;
  Wt[n * K + k] = f2bf(v);
}

// -------------------------------------------------- K1: layernorm + cast ----
__global__ __launch_bounds__(256)
void lnorm_kernel(const float* __restrict__ x, const float* __restrict__ g,
                  const float* __restrict__ beta, ushort_t* __restrict__ xbf) {
  int lane = threadIdx.x & 31;
  int wave = threadIdx.x >> 5;
  int row  = blockIdx.x * (blockDim.x >> 5) + wave;
  if (row >= ROWS) return;
  const float4* xr = (const float4*)(x + row * F_DIM);
  float4 v0 = xr[lane * 2 + 0];
  float4 v1 = xr[lane * 2 + 1];
  float vv[8] = {v0.x, v0.y, v0.z, v0.w, v1.x, v1.y, v1.z, v1.w};
  float s = 0.f, ss = 0.f;
#pragma unroll
  for (int i = 0; i < 8; ++i) { s += vv[i]; ss += vv[i] * vv[i]; }
#pragma unroll
  for (int m = 16; m; m >>= 1) {
    s  += __shfl_xor(s,  m, 32);
    ss += __shfl_xor(ss, m, 32);
  }
  float mu  = s / (float)F_DIM;
  float var = ss / (float)F_DIM - mu * mu;
  float rs  = rsqrtf(var + EPS_LN);
  int c0 = lane * 8;
  union { ushort_t u[8]; uint4 q; } pk;
#pragma unroll
  for (int i = 0; i < 8; ++i)
    pk.u[i] = f2bf((vv[i] - mu) * rs * g[c0 + i] + beta[c0 + i]);
  *(uint4*)(xbf + row * F_DIM + c0) = pk.q;
}

// ------------------------------------------- K2a: wide WMMA GEMM (N=256) ----
// Block = 4 waves = one 16-row M-tile x 256 cols. The 16x32 A-slab is staged
// into LDS once per K-step (async-to-LDS when available) and reused by all 4
// waves; each wave computes a 16x64 tile (4 WMMA accumulators).
// mode 0: bf16 row-major (q,k)   mode 1: f32 sigmoid (gate)
// mode 3: bf16 [b][h][c][L] (v)  mode 4: f32 row-major (final out)
__global__ __launch_bounds__(128)
void gemm_wide_kernel(const ushort_t* __restrict__ A, const ushort_t* __restrict__ Bt,
                      const float* __restrict__ bias, void* __restrict__ outp,
                      float scale, int mode) {
  __shared__ ushort_t shA[16 * 32];
  int t    = threadIdx.x;
  int lane = t & 31;
  int wave = t >> 5;
  int m0   = blockIdx.x << 4;
  int n0   = wave << 6;
  int half = lane >> 4, l16 = lane & 15;
  int kA = half ? 16 : 0;   // A: K run  [kA,kA+16)
  int kB = half ? 16 : 0;   // B: K run  [kB,kB+16)

  const ushort_t* bptr0 = Bt + (n0 + l16) * 256 + kB;
  v8f acc[4] = {};

  int srow = t >> 2, scol = (t & 3) * 8;                  // threads 0..63 stage
  const ushort_t* gsrc = A + (m0 + srow) * 256 + scol;
  ushort_t* ldst = shA + srow * 32 + scol;

  for (int k0 = 0; k0 < 256; k0 += 32) {
    __syncthreads();                        // previous-step LDS reads done
    if (t < 64) {
#if HAVE_ASYNC
      __builtin_amdgcn_global_load_async_to_lds_b128(
          to_global(gsrc + k0), to_lds(ldst), 0, 0);
#else
      *(uint4*)ldst = *(const uint4*)(gsrc + k0);
#endif
    }
#if HAVE_ASYNC
#if __has_builtin(__builtin_amdgcn_s_wait_asynccnt)
    __builtin_amdgcn_s_wait_asynccnt(0);
#else
    asm volatile("s_wait_asynccnt 0x0" ::: "memory");
#endif
#endif
    __syncthreads();                        // LDS slab visible to all waves
    const ushort_t* ap = shA + l16 * 32 + kA;
    v16bf af = make_frag(ap, ap + 8);
#pragma unroll
    for (int j = 0; j < 4; ++j) {
      const ushort_t* bp = bptr0 + j * 16 * 256 + k0;
      __builtin_prefetch((const void*)(bp + 128), 0, 1);
      v16bf bf = make_frag(bp, bp + 8);
      acc[j] = wmma_bf16(af, bf, acc[j]);   // A-fragment reused 4x
    }
  }

#pragma unroll
  for (int j = 0; j < 4; ++j) {
    int col = n0 + j * 16 + l16;
    float bv = bias[col];
#pragma unroll
    for (int r = 0; r < 8; ++r) {
      int mrow = m0 + r + (half << 3);
      float d = (acc[j][r] + bv) * scale;
      if (mode == 0) {
        ((ushort_t*)outp)[mrow * 256 + col] = f2bf(d);
      } else if (mode == 1) {
        ((float*)outp)[mrow * 256 + col] = 1.f / (1.f + __expf(-d));
      } else if (mode == 3) {
        int lseq = mrow >> 3, bb = mrow & 7;   // row = l*B + b, B=8
        int h = col >> 5, c = col & 31;
        ((ushort_t*)outp)[(((bb * H_DIM + h) * C_DIM + c) * L_DIM) + lseq] = f2bf(d);
      } else {
        ((float*)outp)[mrow * 256 + col] = d;
      }
    }
  }
}

// -------------------------------------- K2b: narrow WMMA GEMM (bias, N=8) ---
__global__ __launch_bounds__(128)
void gemm_kernel(const ushort_t* __restrict__ A, const ushort_t* __restrict__ Bt,
                 const float* __restrict__ bias, float* __restrict__ outp,
                 int M, int N) {
  int lane = threadIdx.x & 31;
  int wave = threadIdx.x >> 5;
  int tile = blockIdx.x * (blockDim.x >> 5) + wave;   // Npad=16 -> NT=1
  if (tile >= (M >> 4)) return;
  int m0 = tile << 4;
  int half = lane >> 4, l16 = lane & 15;
  int kA = half ? 16 : 0;
  int kB = half ? 16 : 0;

  const ushort_t* aptr = A + (m0 + l16) * 256;
  const ushort_t* bptr = Bt + l16 * 256 + kB;
  v8f acc = {};
  for (int k0 = 0; k0 < 256; k0 += 32) {
    v16bf af = make_frag(aptr + k0 + kA, aptr + k0 + kA + 8);
    v16bf bf = make_frag(bptr + k0, bptr + k0 + 8);
    acc = wmma_bf16(af, bf, acc);
  }
  int col = l16;
  float bv = (col < N) ? bias[col] : 0.f;
#pragma unroll
  for (int r = 0; r < 8; ++r) {
    int mrow = m0 + r + (half << 3);
    if (col < N) outp[mrow * N + col] = acc[r] + bv;
  }
}

// ------------------------------------------------- K3: flash attention ------
__global__ __launch_bounds__(128)
void attn_kernel(const ushort_t* __restrict__ qbf, const ushort_t* __restrict__ kbf,
                 const ushort_t* __restrict__ vt, const float* __restrict__ biasb,
                 const float* __restrict__ gate, ushort_t* __restrict__ abf) {
  __shared__ ushort_t plds[4 * 16 * 32];
  int lane = threadIdx.x & 31;
  int wave = threadIdx.x >> 5;
  int tile = blockIdx.x * 4 + wave;
  if (tile >= (L_DIM / 16) * B_DIM * H_DIM) return;
  int q0 = (tile >> 6) << 4;          // / (B*H) * 16
  int rem = tile & 63;
  int b  = rem >> 3;
  int hh = rem & 7;
  int half = lane >> 4, l16 = lane & 15;
  int kA = half ? 16 : 0;
  int kB = half ? 16 : 0;
  ushort_t* myp = plds + wave * (16 * 32);

  const ushort_t* qptr = qbf + ((q0 + l16) * B_DIM + b) * HC + hh * C_DIM + kA;
  v16bf qf = make_frag(qptr, qptr + 8);

  v8f acc0 = {}, acc1 = {};
  float mrow[8], lrow[8];
#pragma unroll
  for (int r = 0; r < 8; ++r) { mrow[r] = -1e30f; lrow[r] = 0.f; }

  const ushort_t* vbase = vt + (b * H_DIM + hh) * C_DIM * L_DIM;

  for (int kt = 0; kt < L_DIM / 32; ++kt) {
    int key0 = kt * 32;
    const ushort_t* kp0 = kbf + ((key0 + l16) * B_DIM + b) * HC + hh * C_DIM + kB;
    const ushort_t* kp1 = kbf + ((key0 + 16 + l16) * B_DIM + b) * HC + hh * C_DIM + kB;
    v16bf kf0 = make_frag(kp0, kp0 + 8);
    v16bf kf1 = make_frag(kp1, kp1 + 8);
    v8f z = {};
    v8f s0 = wmma_bf16(qf, kf0, z);
    v8f s1 = wmma_bf16(qf, kf1, z);
    float b0 = biasb[((key0 + l16) * B_DIM + b) * H_DIM + hh];
    float b1 = biasb[((key0 + 16 + l16) * B_DIM + b) * H_DIM + hh];
#pragma unroll
    for (int r = 0; r < 8; ++r) {
      float t0 = s0[r] + b0;
      float t1 = s1[r] + b1;
      float tm = halfmax(fmaxf(t0, t1));
      float mn = fmaxf(mrow[r], tm);
      float sc = __expf(mrow[r] - mn);
      float p0 = __expf(t0 - mn);
      float p1 = __expf(t1 - mn);
      lrow[r] = lrow[r] * sc + halfsum(p0 + p1);
      mrow[r] = mn;
      acc0[r] *= sc;
      acc1[r] *= sc;
      int prow = r + (half << 3);
      myp[prow * 32 + l16]      = f2bf(p0);
      myp[prow * 32 + 16 + l16] = f2bf(p1);
    }
    __syncthreads();
    const ushort_t* pp = myp + l16 * 32 + kA;
    v16bf pf = make_frag(pp, pp + 8);
    const ushort_t* vp0 = vbase + (l16)      * L_DIM + key0 + kB;
    const ushort_t* vp1 = vbase + (16 + l16) * L_DIM + key0 + kB;
    v16bf vf0 = make_frag(vp0, vp0 + 8);
    v16bf vf1 = make_frag(vp1, vp1 + 8);
    acc0 = wmma_bf16(pf, vf0, acc0);
    acc1 = wmma_bf16(pf, vf1, acc1);
    __syncthreads();
  }

#pragma unroll
  for (int r = 0; r < 8; ++r) {
    int qrow = q0 + r + (half << 3);
    float inv = 1.f / lrow[r];
    int base = (qrow * B_DIM + b) * HC + hh * C_DIM;
    float g0 = gate[base + l16];
    float g1 = gate[base + 16 + l16];
    abf[base + l16]      = f2bf(acc0[r] * inv * g0);
    abf[base + 16 + l16] = f2bf(acc1[r] * inv * g1);
  }
}

// -------------------------------------------------------------- launcher ----

extern "C" void kernel_launch(void* const* d_in, const int* in_sizes, int n_in,
                              void* d_out, int out_size, void* d_ws, size_t ws_size,
                              hipStream_t stream) {
  (void)in_sizes; (void)n_in; (void)out_size; (void)ws_size;
  const float* features = (const float*)d_in[0];
  const float* ln_g = (const float*)d_in[1];
  const float* ln_b = (const float*)d_in[2];
  const float* Wq = (const float*)d_in[3];
  const float* bq = (const float*)d_in[4];
  const float* Wk = (const float*)d_in[5];
  const float* bk = (const float*)d_in[6];
  const float* Wv = (const float*)d_in[7];
  const float* bv = (const float*)d_in[8];
  const float* Wb = (const float*)d_in[9];
  const float* bb = (const float*)d_in[10];
  const float* Wg = (const float*)d_in[11];
  const float* bg = (const float*)d_in[12];
  const float* Wo = (const float*)d_in[13];
  const float* bo = (const float*)d_in[14];
  float* out = (float*)d_out;

  char* ws = (char*)d_ws;
  size_t off = 0;
  auto alloc = [&](size_t bytes) {
    size_t o = off;
    off += (bytes + 255) & ~(size_t)255;
    return o;
  };
  ushort_t* xbf   = (ushort_t*)(ws + alloc((size_t)ROWS * F_DIM * 2));
  ushort_t* qbf   = (ushort_t*)(ws + alloc((size_t)ROWS * HC * 2));
  ushort_t* kbf   = (ushort_t*)(ws + alloc((size_t)ROWS * HC * 2));
  ushort_t* vt    = (ushort_t*)(ws + alloc((size_t)B_DIM * H_DIM * C_DIM * L_DIM * 2));
  float*    gatef = (float*)   (ws + alloc((size_t)ROWS * HC * 4));
  float*    biasb = (float*)   (ws + alloc((size_t)ROWS * H_DIM * 4));
  ushort_t* abf   = (ushort_t*)(ws + alloc((size_t)ROWS * HC * 2));
  ushort_t* Wqt   = (ushort_t*)(ws + alloc((size_t)HC * F_DIM * 2));
  ushort_t* Wkt   = (ushort_t*)(ws + alloc((size_t)HC * F_DIM * 2));
  ushort_t* Wvt   = (ushort_t*)(ws + alloc((size_t)HC * F_DIM * 2));
  ushort_t* Wgt   = (ushort_t*)(ws + alloc((size_t)HC * F_DIM * 2));
  ushort_t* Wot   = (ushort_t*)(ws + alloc((size_t)F_DIM * HC * 2));
  ushort_t* Wbt   = (ushort_t*)(ws + alloc((size_t)16 * F_DIM * 2));

  // K0: transpose + bf16-cast weights (pad Wb's N=8 -> 16 with zeros)
  wtrans_kernel<<<(256 * 256 + 255) / 256, 256, 0, stream>>>(Wq, Wqt, F_DIM, HC, HC);
  wtrans_kernel<<<(256 * 256 + 255) / 256, 256, 0, stream>>>(Wk, Wkt, F_DIM, HC, HC);
  wtrans_kernel<<<(256 * 256 + 255) / 256, 256, 0, stream>>>(Wv, Wvt, F_DIM, HC, HC);
  wtrans_kernel<<<(256 * 256 + 255) / 256, 256, 0, stream>>>(Wg, Wgt, F_DIM, HC, HC);
  wtrans_kernel<<<(256 * 256 + 255) / 256, 256, 0, stream>>>(Wo, Wot, HC, F_DIM, F_DIM);
  wtrans_kernel<<<(16 * 256 + 255) / 256, 256, 0, stream>>>(Wb, Wbt, F_DIM, H_DIM, 16);

  // K1: layernorm
  lnorm_kernel<<<ROWS / 8, 256, 0, stream>>>(features, ln_g, ln_b, xbf);

  // K2: projections (wide GEMM: 512 blocks, one M-tile each)
  const float sq_c = 0.17677669529663687f;  // 1/sqrt(32)
  gemm_wide_kernel<<<ROWS / 16, 128, 0, stream>>>(xbf, Wqt, bq, qbf, sq_c, 0);
  gemm_wide_kernel<<<ROWS / 16, 128, 0, stream>>>(xbf, Wkt, bk, kbf, 1.0f, 0);
  gemm_wide_kernel<<<ROWS / 16, 128, 0, stream>>>(xbf, Wvt, bv, vt, 1.0f, 3);
  gemm_wide_kernel<<<ROWS / 16, 128, 0, stream>>>(xbf, Wgt, bg, gatef, 1.0f, 1);
  gemm_kernel<<<(ROWS / 16 + 3) / 4, 128, 0, stream>>>(xbf, Wbt, bb, biasb, ROWS, H_DIM);

  // K3: attention (4096 wave-tiles, 4 per block)
  attn_kernel<<<(L_DIM / 16) * B_DIM * H_DIM / 4, 128, 0, stream>>>(qbf, kbf, vt, biasb, gatef, abf);

  // K4: output projection -> f32 d_out
  gemm_wide_kernel<<<ROWS / 16, 128, 0, stream>>>(abf, Wot, bo, out, 1.0f, 4);
}
